// ResidualGNNDenoiser_80298708566763
// MI455X (gfx1250) — matrix operand order repaired and checked by reading, not compiled, not run on hardware
//
#include <hip/hip_runtime.h>
#include <math.h>

typedef _Float16 v8h  __attribute__((ext_vector_type(8)));
typedef _Float16 v16h __attribute__((ext_vector_type(16)));
typedef float    v8f  __attribute__((ext_vector_type(8)));

typedef int v4i_ __attribute__((vector_size(16)));
typedef __attribute__((address_space(1))) v4i_ gv4i;   // global int4
typedef __attribute__((address_space(3))) v4i_ lv4i;   // LDS int4

#define LDW 136   // padded row stride (halfs): 272B rows keep 16B alignment, kill LDS bank conflicts
#define WPACK_HALFS (128 * LDW)          // packed weight buffer size (halfs)
#define WPACK_CHUNKS ((WPACK_HALFS * 2) / 16)  // 16B chunks = 2176

#if __has_builtin(__builtin_amdgcn_global_load_async_to_lds_b128)
#define HAVE_ASYNC_LDS 1
#else
#define HAVE_ASYNC_LDS 0
#endif

__device__ __forceinline__ float gelu_exact(float v) {
    return 0.5f * v * (1.0f + erff(v * 0.70710678118654752f));
}

__device__ __forceinline__ void wait_asynccnt0() {
#if __has_builtin(__builtin_amdgcn_s_wait_asynccnt)
    __builtin_amdgcn_s_wait_asynccnt(0);
#else
    asm volatile("s_wait_asynccnt 0x0" ::: "memory");
#endif
}

// ---------------------------------------------------------------------------
// Fused node embedding: Fourier(t) -> silu MLP -> temb; + x@pxW + c@pcW + biases
// One 128-thread block per node. Produces H in f16 (WMMA consumes f16 anyway).
// ---------------------------------------------------------------------------
__global__ __launch_bounds__(128) void node_embed_kernel(
    const float* __restrict__ x, const float* __restrict__ ts,
    const float* __restrict__ cond, const float* __restrict__ fw,
    const float* __restrict__ tW1, const float* __restrict__ tb1,
    const float* __restrict__ tW2, const float* __restrict__ tb2,
    const float* __restrict__ pxW, const float* __restrict__ pxb,
    const float* __restrict__ pcW, const float* __restrict__ pcb,
    _Float16* __restrict__ Hh, int n, int rep)
{
    __shared__ float four[17];
    __shared__ float xrow[64];
    __shared__ float crow[32];
    __shared__ float hid[128];

    const int node = blockIdx.x;
    const int t = threadIdx.x;
    const float tv = ts[node];

    if (t < 8) {
        four[1 + t] = __sinf(tv * fw[t] * 6.28318530717958648f);
    } else if (t < 16) {
        four[9 + (t - 8)] = __cosf(tv * fw[t - 8] * 6.28318530717958648f);
    } else if (t == 16) {
        four[0] = tv;
    }
    if (t < 64) xrow[t] = x[(size_t)node * 64 + t];
    if (t < 32) crow[t] = cond[(size_t)(node / rep) * 32 + t];
    __syncthreads();

    float hsum = tb1[t];
#pragma unroll
    for (int i = 0; i < 17; ++i) hsum += four[i] * tW1[i * 128 + t];
    hid[t] = hsum / (1.0f + __expf(-hsum));   // silu
    __syncthreads();

    float acc = tb2[t] + pxb[t] + pcb[t];
    for (int k = 0; k < 128; ++k) acc += hid[k] * tW2[k * 128 + t];
#pragma unroll 4
    for (int i = 0; i < 64; ++i) acc += xrow[i] * pxW[i * 128 + t];
#pragma unroll 4
    for (int i = 0; i < 32; ++i) acc += crow[i] * pcW[i * 128 + t];
    Hh[(size_t)node * 128 + t] = (_Float16)acc;
}

// ---------------------------------------------------------------------------
// Pack a 128x128 f32 weight matrix -> transposed, padded f16 buffer
// Wp[col*LDW + k] = (f16)W[k*128 + col]; makes GEMM staging a raw byte copy.
// ---------------------------------------------------------------------------
__global__ __launch_bounds__(256) void pack_w_kernel(
    const float* __restrict__ W, _Float16* __restrict__ Wp)
{
    const int i = blockIdx.x * 256 + threadIdx.x;
    if (i >= 128 * 128) return;
    const int k = i >> 7;
    const int c = i & 127;
    Wp[c * LDW + k] = (_Float16)W[i];
}

// ---------------------------------------------------------------------------
// WMMA GEMM: C[n x 128] = A[n x 128](f16) @ W[128 x 128] (pre-packed f16)
// 256 threads = 8 waves; block covers 128 rows, each wave a 16x128 strip.
// Packed weights staged to LDS via GLOBAL_LOAD_ASYNC_TO_LDS_B128 (ASYNCcnt).
// ---------------------------------------------------------------------------
__global__ __launch_bounds__(256) void gemm128_wmma_kernel(
    const _Float16* __restrict__ A, const _Float16* __restrict__ Wp,
    float* __restrict__ C, int n)
{
    __shared__ _Float16 sW[WPACK_HALFS];   // sW[col*LDW + k]

    const int tid  = threadIdx.x;
    const int wave = tid >> 5;
    const int lane = tid & 31;
    const int mrow = lane & 15;
    const int halfsel = lane >> 4;       // 0: lanes 0-15, 1: lanes 16-31
    const int rbase = blockIdx.x * 128 + wave * 16;

    // Stage packed weights (raw 16B chunks) global -> LDS.
    for (int c = tid; c < WPACK_CHUNKS; c += 256) {
#if HAVE_ASYNC_LDS
        __builtin_amdgcn_global_load_async_to_lds_b128(
            (gv4i*)(Wp + c * 8), (lv4i*)(sW + c * 8), 0, 0);
#else
        *(v8h*)(sW + c * 8) = *(const v8h*)(Wp + c * 8);
#endif
    }
#if HAVE_ASYNC_LDS
    wait_asynccnt0();
#endif
    __syncthreads();

    v8f acc[8] = {};

    // A row this lane feeds (clamped; extra rows computed but never stored)
    int arow = rbase + mrow;
    if (arow >= n) arow = n - 1;
    const _Float16* __restrict__ ap = A + (size_t)arow * 128;
    __builtin_prefetch(ap, 0, 0);   // global_prefetch_b8 into the cache path

#pragma unroll
    for (int k0 = 0; k0 < 128; k0 += 32) {
        // A fragment: lane<16 -> K {k0..k0+7, k0+16..k0+23}; lane>=16 -> +8
        const int kb = k0 + halfsel * 8;
        const v8h alo = *(const v8h*)(ap + kb);
        const v8h ahi = *(const v8h*)(ap + kb + 16);
        const v16h a = __builtin_shufflevector(alo, ahi,
            0, 1, 2, 3, 4, 5, 6, 7, 8, 9, 10, 11, 12, 13, 14, 15);
        // B fragments: lane<16 -> K {k0..k0+7, k0+16..k0+23}; lane>=16 -> +8
        const int kbB = k0 + halfsel * 8;
#pragma unroll
        for (int t = 0; t < 8; ++t) {
            const _Float16* bCol = &sW[(t * 16 + mrow) * LDW];
            const v8h blo = *(const v8h*)(bCol + kbB);
            const v8h bhi = *(const v8h*)(bCol + kbB + 16);
            const v16h b = __builtin_shufflevector(blo, bhi,
                0, 1, 2, 3, 4, 5, 6, 7, 8, 9, 10, 11, 12, 13, 14, 15);
            acc[t] = __builtin_amdgcn_wmma_f32_16x16x32_f16(
                false, a, false, b, (short)0, acc[t], false, false);
        }
    }

    // C/D layout: VGPR r -> M = r + 8*halfsel, N = t*16 + mrow
#pragma unroll
    for (int t = 0; t < 8; ++t) {
        const int col = t * 16 + mrow;
#pragma unroll
        for (int r = 0; r < 8; ++r) {
            const int grow = rbase + halfsel * 8 + r;
            if (grow < n) C[(size_t)grow * 128 + col] = acc[t][r];
        }
    }
}

// ---------------------------------------------------------------------------
// a_src/a_dst: per-node dot(xl_row, attS/attD). One wave per node.
// ---------------------------------------------------------------------------
__global__ __launch_bounds__(256) void att_kernel(
    const float* __restrict__ XL, const float* __restrict__ vS,
    const float* __restrict__ vD, float* __restrict__ as_,
    float* __restrict__ ad_, int n)
{
    const int node = blockIdx.x * 8 + (threadIdx.x >> 5);
    const int lane = threadIdx.x & 31;
    if (node >= n) return;
    const float4 xv = ((const float4*)(XL + (size_t)node * 128))[lane];
    const float4 sv = ((const float4*)vS)[lane];
    const float4 dv = ((const float4*)vD)[lane];
    float ss = xv.x * sv.x + xv.y * sv.y + xv.z * sv.z + xv.w * sv.w;
    float dd = xv.x * dv.x + xv.y * dv.y + xv.z * dv.z + xv.w * dv.w;
#pragma unroll
    for (int off = 16; off > 0; off >>= 1) {
        ss += __shfl_down(ss, off);
        dd += __shfl_down(dd, off);
    }
    if (lane == 0) { as_[node] = ss; ad_[node] = dd; }
}

// ---------------------------------------------------------------------------
// init segment buffers: OUT=0, m=-inf, den=0
// ---------------------------------------------------------------------------
__global__ __launch_bounds__(256) void seg_init_kernel(
    float* __restrict__ OUT, float* __restrict__ m, float* __restrict__ den, int n)
{
    const int idx = blockIdx.x * 256 + threadIdx.x;
    if (idx < n * 128) OUT[idx] = 0.0f;
    if (idx < n) { m[idx] = -3.0e38f; den[idx] = 0.0f; }
}

// ---------------------------------------------------------------------------
// edge pass 1: e = leaky_relu(a_src[s] + a_dst[d]); segment max via f32 atomic
// ---------------------------------------------------------------------------
__global__ __launch_bounds__(256) void edge_pass1_kernel(
    const int* __restrict__ srcI, const int* __restrict__ dstI,
    const float* __restrict__ as_, const float* __restrict__ ad_,
    float* __restrict__ el, float* __restrict__ m, int Eg, int ET)
{
    const int i = blockIdx.x * 256 + threadIdx.x;
    if (i >= ET) return;
    const int s = (i < Eg) ? srcI[i] : (i - Eg);
    const int d = (i < Eg) ? dstI[i] : (i - Eg);
    const float v = as_[s] + ad_[d];
    const float e = (v > 0.0f) ? v : 0.2f * v;
    el[i] = e;
    atomicMax(&m[d], e);
}

// ---------------------------------------------------------------------------
// edge pass 2: den[d] += exp(e - m[d])
// ---------------------------------------------------------------------------
__global__ __launch_bounds__(256) void edge_pass2_kernel(
    const int* __restrict__ dstI, const float* __restrict__ el,
    const float* __restrict__ m, float* __restrict__ den, int Eg, int ET)
{
    const int i = blockIdx.x * 256 + threadIdx.x;
    if (i >= ET) return;
    const int d = (i < Eg) ? dstI[i] : (i - Eg);
    atomicAdd(&den[d], __expf(el[i] - m[d]));
}

// ---------------------------------------------------------------------------
// edge pass 3: OUT[d] += alpha * XL[s];  one wave per edge, 4 ch per lane
// ---------------------------------------------------------------------------
__global__ __launch_bounds__(256) void edge_pass3_kernel(
    const int* __restrict__ srcI, const int* __restrict__ dstI,
    const float* __restrict__ el, const float* __restrict__ m,
    const float* __restrict__ den, const float* __restrict__ XL,
    float* __restrict__ OUT, int Eg, int ET)
{
    const int i = blockIdx.x * 8 + (threadIdx.x >> 5);
    const int lane = threadIdx.x & 31;
    if (i >= ET) return;
    const int s = (i < Eg) ? srcI[i] : (i - Eg);
    const int d = (i < Eg) ? dstI[i] : (i - Eg);
    const float alpha = __expf(el[i] - m[d]) / den[d];
    const float4 xv = ((const float4*)(XL + (size_t)s * 128))[lane];
    float* o = OUT + (size_t)d * 128 + lane * 4;
    atomicAdd(o + 0, alpha * xv.x);
    atomicAdd(o + 1, alpha * xv.y);
    atomicAdd(o + 2, alpha * xv.z);
    atomicAdd(o + 3, alpha * xv.w);
}

// ---------------------------------------------------------------------------
// H = gelu(OUT + gbias + RES), written as f16 for the next WMMA stage
// ---------------------------------------------------------------------------
__global__ __launch_bounds__(256) void combine_gelu_kernel(
    const float* __restrict__ OUT, const float* __restrict__ RES,
    const float* __restrict__ gb, _Float16* __restrict__ Hh, int n)
{
    const int idx = blockIdx.x * 256 + threadIdx.x;
    if (idx >= n * 128) return;
    const float v = OUT[idx] + gb[idx & 127] + RES[idx];
    Hh[idx] = (_Float16)gelu_exact(v);
}

// ---------------------------------------------------------------------------
// Z = gelu(Z + db1), in place (f32)
// ---------------------------------------------------------------------------
__global__ __launch_bounds__(256) void bias_gelu_kernel(
    float* __restrict__ Z, const float* __restrict__ b, int n)
{
    const int idx = blockIdx.x * 256 + threadIdx.x;
    if (idx >= n * 128) return;
    Z[idx] = gelu_exact(Z[idx] + b[idx & 127]);
}

// ---------------------------------------------------------------------------
// out[i,0:2] = Z[i] @ dW2 + db2;  one wave per node
// ---------------------------------------------------------------------------
__global__ __launch_bounds__(256) void final_head_kernel(
    const float* __restrict__ Z, const float* __restrict__ dW2,
    const float* __restrict__ db2, float* __restrict__ out, int n)
{
    const int node = blockIdx.x * 8 + (threadIdx.x >> 5);
    const int lane = threadIdx.x & 31;
    if (node >= n) return;
    const float4 z = ((const float4*)(Z + (size_t)node * 128))[lane];
    const int c0 = lane * 4;
    float s0 = z.x * dW2[(c0 + 0) * 2] + z.y * dW2[(c0 + 1) * 2] +
               z.z * dW2[(c0 + 2) * 2] + z.w * dW2[(c0 + 3) * 2];
    float s1 = z.x * dW2[(c0 + 0) * 2 + 1] + z.y * dW2[(c0 + 1) * 2 + 1] +
               z.z * dW2[(c0 + 2) * 2 + 1] + z.w * dW2[(c0 + 3) * 2 + 1];
#pragma unroll
    for (int off = 16; off > 0; off >>= 1) {
        s0 += __shfl_down(s0, off);
        s1 += __shfl_down(s1, off);
    }
    if (lane == 0) {
        out[(size_t)node * 2 + 0] = s0 + db2[0];
        out[(size_t)node * 2 + 1] = s1 + db2[1];
    }
}

// ---------------------------------------------------------------------------
extern "C" void kernel_launch(void* const* d_in, const int* in_sizes, int n_in,
                              void* d_out, int out_size, void* d_ws, size_t ws_size,
                              hipStream_t stream)
{
    (void)n_in; (void)out_size; (void)ws_size;

    const float* x    = (const float*)d_in[0];
    const int*   adj  = (const int*)d_in[1];
    const float* ts   = (const float*)d_in[3];
    const float* cond = (const float*)d_in[4];
    const float* fw   = (const float*)d_in[5];
    const float* tW1  = (const float*)d_in[6];
    const float* tb1  = (const float*)d_in[7];
    const float* tW2  = (const float*)d_in[8];
    const float* tb2  = (const float*)d_in[9];
    const float* pxW  = (const float*)d_in[10];
    const float* pxb  = (const float*)d_in[11];
    const float* pcW  = (const float*)d_in[12];
    const float* pcb  = (const float*)d_in[13];
    const float* linW = (const float*)d_in[14];
    const float* attS = (const float*)d_in[15];
    const float* attD = (const float*)d_in[16];
    const float* gbias= (const float*)d_in[17];
    const float* resW = (const float*)d_in[18];
    const float* dW1  = (const float*)d_in[19];
    const float* db1  = (const float*)d_in[20];
    const float* dW2  = (const float*)d_in[21];
    const float* db2  = (const float*)d_in[22];
    float* out = (float*)d_out;

    const int N   = in_sizes[3];            // 50000
    const int Eg  = in_sizes[1] / 2;        // 600000
    const int Bc  = in_sizes[4] / 32;       // 50
    const int L   = in_sizes[14] / (128 * 128); // 3
    const int ET  = Eg + N;                 // edges incl. self loops
    const int rep = N / Bc;                 // 1000

    // workspace layout (16B-aligned sections)
    float* XL  = (float*)d_ws;               // N*128 f32 (xl / res / z)
    float* OUT = XL  + (size_t)N * 128;      // N*128 f32
    float* asr = OUT + (size_t)N * 128;      // N
    float* adt = asr + N;                    // N
    float* m   = adt + N;                    // N
    float* den = m   + N;                    // N
    float* el  = den + N;                    // ET
    _Float16* Hh = (_Float16*)(el + ((ET + 63) & ~63));   // N*128 f16
    _Float16* Wp = Hh + (size_t)N * 128;                  // packed weights (f16)

    const int* srcI = adj;
    const int* dstI = adj + Eg;

    const int gemmBlocks = (N + 127) / 128;
    const int nkBlocks   = (N * 128 + 255) / 256;
    const int eBlocks    = (ET + 255) / 256;
    const int e8Blocks   = (ET + 7) / 8;
    const int nwBlocks   = (N + 7) / 8;
    const int pwBlocks   = (128 * 128 + 255) / 256;

    node_embed_kernel<<<N, 128, 0, stream>>>(x, ts, cond, fw, tW1, tb1, tW2, tb2,
                                             pxW, pxb, pcW, pcb, Hh, N, rep);

    for (int l = 0; l < L; ++l) {
        pack_w_kernel<<<pwBlocks, 256, 0, stream>>>(linW + (size_t)l * 16384, Wp);
        gemm128_wmma_kernel<<<gemmBlocks, 256, 0, stream>>>(Hh, Wp, XL, N);
        att_kernel<<<nwBlocks, 256, 0, stream>>>(
            XL, attS + l * 128, attD + l * 128, asr, adt, N);
        seg_init_kernel<<<nkBlocks, 256, 0, stream>>>(OUT, m, den, N);
        edge_pass1_kernel<<<eBlocks, 256, 0, stream>>>(srcI, dstI, asr, adt, el, m, Eg, ET);
        edge_pass2_kernel<<<eBlocks, 256, 0, stream>>>(dstI, el, m, den, Eg, ET);
        edge_pass3_kernel<<<e8Blocks, 256, 0, stream>>>(srcI, dstI, el, m, den, XL, OUT, Eg, ET);
        // residual GEMM into XL (XL dead after pass3)
        pack_w_kernel<<<pwBlocks, 256, 0, stream>>>(resW + (size_t)l * 16384, Wp);
        gemm128_wmma_kernel<<<gemmBlocks, 256, 0, stream>>>(Hh, Wp, XL, N);
        combine_gelu_kernel<<<nkBlocks, 256, 0, stream>>>(OUT, XL, gbias + l * 128, Hh, N);
    }

    pack_w_kernel<<<pwBlocks, 256, 0, stream>>>(dW1, Wp);
    gemm128_wmma_kernel<<<gemmBlocks, 256, 0, stream>>>(Hh, Wp, XL, N);
    bias_gelu_kernel<<<nkBlocks, 256, 0, stream>>>(XL, db1, N);
    final_head_kernel<<<nwBlocks, 256, 0, stream>>>(XL, dW2, db2, out, N);
}
